// TorchaoQuantLinearInt8_7078106103817
// MI455X (gfx1250) — 8-bit matrix operand order repaired and checked by reading, compiled, not run on hardware
//
#include <hip/hip_runtime.h>
#include <hip/hip_bf16.h>
#include <stdint.h>

typedef int v8i __attribute__((ext_vector_type(8)));
typedef int v4i __attribute__((ext_vector_type(4)));
typedef int v2i __attribute__((ext_vector_type(2)));

typedef int v4i_g __attribute__((vector_size(16)));
typedef __attribute__((address_space(1))) v4i_g* gv4p;
typedef __attribute__((address_space(3))) v4i_g* lv4p;

#define T_DIM 4096
#define K_DIM 4096
#define N_DIM 11008

#define BM 128
#define BN 256
#define BK 64
#define LDA 80
#define LDB 80
#define NKITER (K_DIM / BK)

#if __has_builtin(__builtin_amdgcn_global_load_async_to_lds_b128)
#define HAS_ASYNC 1
#endif

__device__ __forceinline__ void cp16_to_lds(const void* g, void* l) {
#ifdef HAS_ASYNC
  __builtin_amdgcn_global_load_async_to_lds_b128((gv4p)g, (lv4p)l, 0, 0);
#else
  *(v4i*)l = *(const v4i*)g;
#endif
}

__device__ __forceinline__ void wait_async_all() {
#ifdef HAS_ASYNC
#if __has_builtin(__builtin_amdgcn_s_wait_asynccnt)
  __builtin_amdgcn_s_wait_asynccnt(0);
#else
  asm volatile("s_wait_asynccnt 0" ::: "memory");
#endif
#endif
}

__global__ __launch_bounds__(256, 1)
void quant_rows(const __hip_bfloat16* __restrict__ x, int8_t* __restrict__ xq,
                float* __restrict__ xs) {
  const int row = blockIdx.x;
  const int tid = threadIdx.x;
  __shared__ float red[8];

  const uint32_t* xr = (const uint32_t*)(x + (size_t)row * K_DIM);
  uint32_t w[8];
  *(v4i*)(w)     = *(const v4i*)(xr + (size_t)tid * 8);
  *(v4i*)(w + 4) = *(const v4i*)(xr + (size_t)tid * 8 + 4);

  float amax = 0.0f;
#pragma unroll
  for (int i = 0; i < 8; ++i) {
    const uint32_t u = w[i];
    amax = fmaxf(amax, fabsf(__uint_as_float(u << 16)));
    amax = fmaxf(amax, fabsf(__uint_as_float(u & 0xffff0000u)));
  }
#pragma unroll
  for (int off = 16; off >= 1; off >>= 1)
    amax = fmaxf(amax, __shfl_xor(amax, off, 32));
  if ((tid & 31) == 0) red[tid >> 5] = amax;
  __syncthreads();
  float m = red[0];
#pragma unroll
  for (int i = 1; i < 8; ++i) m = fmaxf(m, red[i]);
  m = fmaxf(m, 1e-5f);
  const float inv = 127.0f / m;

  uint32_t pk[4];
#pragma unroll
  for (int i = 0; i < 4; ++i) {
    uint32_t r = 0;
#pragma unroll
    for (int j = 0; j < 2; ++j) {
      const uint32_t u = w[i * 2 + j];
      const float f0 = __uint_as_float(u << 16);
      const float f1 = __uint_as_float(u & 0xffff0000u);
      int q0 = (int)rintf(f0 * inv);
      int q1 = (int)rintf(f1 * inv);
      q0 = q0 > 127 ? 127 : (q0 < -127 ? -127 : q0);
      q1 = q1 > 127 ? 127 : (q1 < -127 ? -127 : q1);
      r |= ((uint32_t)(q0 & 0xff)) << (j * 16);
      r |= ((uint32_t)(q1 & 0xff)) << (j * 16 + 8);
    }
    pk[i] = r;
  }
  *(v4i*)(xq + (size_t)row * K_DIM + tid * 16) = *(const v4i*)pk;
  if (tid == 0) xs[row] = m * (1.0f / 127.0f);
}

__global__ __launch_bounds__(256, 1)
void int8_wmma_gemm(const int8_t* __restrict__ xq, const float* __restrict__ xs,
                    const int8_t* __restrict__ wq, const float* __restrict__ wsc,
                    const __hip_bfloat16* __restrict__ bias,
                    __hip_bfloat16* __restrict__ out) {
  __shared__ char ldsA[2][BM * LDA];
  __shared__ char ldsB[2][BN * LDB];

  const int tid  = threadIdx.x;
  const int lane = tid & 31;
  const int wave = tid >> 5;
  const int nBase = blockIdx.x * BN;
  const int tBase = blockIdx.y * BM;
  const int wm = (wave & 1) * 64;
  const int wn = (wave >> 1) * 64;
  const int lm = lane & 15;
  const int hi = lane >> 4;

  const int ra0 = tid >> 2;
  const int ra1 = ra0 + 64;
  const int rb  = tid >> 2;
  const int cc  = (tid & 3) * 16;

  const int8_t* gA = xq + (size_t)tBase * K_DIM;
  const int8_t* gB = wq + (size_t)nBase * K_DIM;

  auto fill = [&](int buf, int kt) {
    const int kOff = kt * BK + cc;
    cp16_to_lds(gA + (size_t)ra0 * K_DIM + kOff, &ldsA[buf][ra0 * LDA + cc]);
    cp16_to_lds(gA + (size_t)ra1 * K_DIM + kOff, &ldsA[buf][ra1 * LDA + cc]);
#pragma unroll
    for (int i = 0; i < 4; ++i) {
      const int r = rb + i * 64;
      cp16_to_lds(gB + (size_t)r * K_DIM + kOff, &ldsB[buf][r * LDB + cc]);
    }
  };

  v8i acc[4][4];
#pragma unroll
  for (int mt = 0; mt < 4; ++mt)
#pragma unroll
    for (int nt = 0; nt < 4; ++nt)
#pragma unroll
      for (int v = 0; v < 8; ++v) acc[mt][nt][v] = 0;

  fill(0, 0);

  for (int kk = 0; kk < NKITER; ++kk) {
    wait_async_all();
    __syncthreads();
    if (kk + 1 < NKITER) fill((kk + 1) & 1, kk + 1);

    const char* As = ldsA[kk & 1];
    const char* Bs = ldsB[kk & 1];

    v8i afrag[4];
#pragma unroll
    for (int mt = 0; mt < 4; ++mt) {
      const char* ab = As + (wm + mt * 16 + lm) * LDA + hi * 16;
      v8i a;
#pragma unroll
      for (int p = 0; p < 4; ++p) {
        v2i d = *(const v2i*)(ab + (p & 1) * 8 + (p >> 1) * 32);
        a[2 * p]     = d.x;
        a[2 * p + 1] = d.y;
      }
      afrag[mt] = a;
    }

    v8i bfrag[4];
#pragma unroll
    for (int nt = 0; nt < 4; ++nt) {
      const char* bb = Bs + (wn + nt * 16 + lm) * LDB + hi * 16;
      const v4i lo = *(const v4i*)(bb);
      const v4i h4 = *(const v4i*)(bb + 32);
      v8i b;
      b[0] = lo.x; b[1] = lo.y; b[2] = lo.z; b[3] = lo.w;
      b[4] = h4.x; b[5] = h4.y; b[6] = h4.z; b[7] = h4.w;
      bfrag[nt] = b;
    }

#pragma unroll
    for (int mt = 0; mt < 4; ++mt)
#pragma unroll
      for (int nt = 0; nt < 4; ++nt)
        acc[mt][nt] = __builtin_amdgcn_wmma_i32_16x16x64_iu8(
            true, afrag[mt], true, bfrag[nt], acc[mt][nt], false, false);
  }

  float wsv[4], bsv[4];
#pragma unroll
  for (int nt = 0; nt < 4; ++nt) {
    const int n = nBase + wn + nt * 16 + lm;
    wsv[nt] = wsc[n];
    bsv[nt] = __bfloat162float(bias[n]);
  }
#pragma unroll
  for (int mt = 0; mt < 4; ++mt) {
#pragma unroll
    for (int v = 0; v < 8; ++v) {
      const int t = tBase + wm + mt * 16 + hi * 8 + v;
      const float sx = xs[t];
#pragma unroll
      for (int nt = 0; nt < 4; ++nt) {
        const int n = nBase + wn + nt * 16 + lm;
        const float val = (float)acc[mt][nt][v] * sx * wsv[nt] + bsv[nt];
        out[(size_t)t * N_DIM + n] = __float2bfloat16(val);
      }
    }
  }
}

extern "C" void kernel_launch(void* const* d_in, const int* in_sizes, int n_in,
                              void* d_out, int out_size, void* d_ws, size_t ws_size,
                              hipStream_t stream) {
  (void)in_sizes; (void)n_in; (void)out_size; (void)ws_size;
  const __hip_bfloat16* x    = (const __hip_bfloat16*)d_in[0];
  const int8_t*         wqp  = (const int8_t*)d_in[1];
  const float*          wsc  = (const float*)d_in[2];
  const __hip_bfloat16* bias = (const __hip_bfloat16*)d_in[3];
  __hip_bfloat16*       out  = (__hip_bfloat16*)d_out;

  int8_t* xq = (int8_t*)d_ws;
  float*  xs = (float*)((char*)d_ws + (size_t)T_DIM * K_DIM);

  quant_rows<<<T_DIM, 256, 0, stream>>>(x, xq, xs);

  dim3 grid(N_DIM / BN, T_DIM / BM);
  int8_wmma_gemm<<<grid, 256, 0, stream>>>(xq, xs, wqp, wsc, bias, out);
}
